// DecoderRNN_24764781428890
// MI455X (gfx1250) — matrix operand order repaired and checked by reading, not compiled, not run on hardware
//
#include <hip/hip_runtime.h>
#include <math.h>

#define BATCH  64
#define SEQ    256
#define EMBED  512
#define HIDDEN 1024
#define GATES  (4 * HIDDEN)
#define KCAT   (EMBED + HIDDEN) /* 1536: [x | h] concat reduction dim */
#define NSPLIT 3                /* K splits of 512 (split 0 = x, 1..2 = h) */
#define NCHUNK 16               /* 512 / 32 chunks per wave */

typedef __bf16 bf16_t;
typedef __attribute__((ext_vector_type(8)))  __bf16 v8bf;
typedef __attribute__((ext_vector_type(16))) __bf16 v16bf;
typedef __attribute__((ext_vector_type(8)))  float  v8f;

__device__ __forceinline__ bf16_t f2bf(float f) {
  unsigned u = __builtin_bit_cast(unsigned, f);
  unsigned r = u + 0x7FFFu + ((u >> 16) & 1u); // round-to-nearest-even
  unsigned short h = (unsigned short)(r >> 16);
  return __builtin_bit_cast(bf16_t, h);
}

__device__ __forceinline__ float sigm(float x) {
  return 1.0f / (1.0f + __expf(-x));
}

#define SHUF16(lo, hi) __builtin_shufflevector(lo, hi, 0, 1, 2, 3, 4, 5, 6, 7, \
                                               8, 9, 10, 11, 12, 13, 14, 15)
#define WMMA_BF16(a, b, c) \
  __builtin_amdgcn_wmma_f32_16x16x32_bf16(false, a, false, b, (short)0, c, false, false)

// Load a 16x32 bf16 A tile chunk: per-lane one contiguous 32B read.
__device__ __forceinline__ v16bf ldA(const bf16_t* pa, int c) {
  return *(const v16bf*)(pa + c * 32);
}
// Load a 32x16 bf16 B tile chunk: per-lane one contiguous 32B read.
__device__ __forceinline__ v16bf ldB(const bf16_t* pb, int c) {
  return *(const v16bf*)(pb + c * 32);
}

// ---------------------------------------------------------------------------
// Setup kernels (one-time bf16 conversion + state init)
// ---------------------------------------------------------------------------
__global__ void k_wcat(const float* __restrict__ Wih, const float* __restrict__ Whh,
                       bf16_t* __restrict__ Wcat) {
  const size_t N = (size_t)GATES * KCAT;
  for (size_t i = (size_t)blockIdx.x * blockDim.x + threadIdx.x; i < N;
       i += (size_t)gridDim.x * blockDim.x) {
    size_t n = i / KCAT, k = i % KCAT;
    float v = (k < EMBED) ? Wih[n * EMBED + k] : Whh[n * HIDDEN + (k - EMBED)];
    Wcat[i] = f2bf(v);
  }
}

__global__ void k_cvt_bf16(const float* __restrict__ src, bf16_t* __restrict__ dst,
                           size_t n) {
  for (size_t i = (size_t)blockIdx.x * blockDim.x + threadIdx.x; i < n;
       i += (size_t)gridDim.x * blockDim.x)
    dst[i] = f2bf(src[i]);
}

__global__ void k_bias(const float* __restrict__ a, const float* __restrict__ b,
                       float* __restrict__ o) {
  for (size_t i = (size_t)blockIdx.x * blockDim.x + threadIdx.x; i < (size_t)GATES;
       i += (size_t)gridDim.x * blockDim.x)
    o[i] = a[i] + b[i];
}

__global__ void k_state_init(bf16_t* __restrict__ h0, float* __restrict__ c) {
  for (size_t i = (size_t)blockIdx.x * blockDim.x + threadIdx.x;
       i < (size_t)BATCH * HIDDEN; i += (size_t)gridDim.x * blockDim.x) {
    h0[i] = f2bf(0.0f);
    c[i]  = 0.0f;
  }
}

// ---------------------------------------------------------------------------
// One LSTM time step, fused, K-split across waves:
//   gates = [x_t | h_{t-1}] @ Wcat^T + bias   (bf16 WMMA, f32 accum)
// Grid: 64 blocks (hidden tiles of 16) x 384 threads (12 waves).
//   wave = ks*4 + mq : mq = batch tile (4 x 16 rows), ks = K split (3 x 512).
// Splits never cross the x/h boundary, so each wave resolves its A base
// pointer ONCE (scalar-uniform), and the fully unrolled 16-chunk K loop uses
// only immediate-offset loads in a 2-deep software pipeline. Partials from
// ks!=0 waves are tree-summed through LDS (32 KB), ks==0 runs the fused
// LSTM-cell epilogue.
// ---------------------------------------------------------------------------
__global__ void __launch_bounds__(384)
k_lstm_step(const bf16_t* __restrict__ Xb, const bf16_t* __restrict__ Wcat,
            const float* __restrict__ bias, const bf16_t* __restrict__ hin,
            bf16_t* __restrict__ hout, float* __restrict__ cbuf,
            bf16_t* __restrict__ Hall, int t) {
  __shared__ float red[8 * 4 * 256]; // [(ks-1)*4+mq][gate][16x16 f32] = 32 KB

  const int lane = threadIdx.x & 31;
  const int wid  = __builtin_amdgcn_readfirstlane((int)threadIdx.x) >> 5; // scalar
  const int mq   = wid & 3;          // batch tile
  const int ks   = wid >> 2;         // K split (0..2)
  const int hlf  = lane >> 4;        // 0: lanes 0-15, 1: lanes 16-31
  const int lrow = lane & 15;
  const int m0   = mq * 16;
  const int n0   = blockIdx.x * 16;  // hidden tile base (1024/16 = 64 blocks)

  // A base pointer: uniform select, done once (no per-iteration math).
  const bf16_t* pa;
  if (ks == 0)
    pa = Xb + (size_t)t * EMBED + (size_t)(m0 + lrow) * ((size_t)SEQ * EMBED) + hlf * 16;
  else
    pa = hin + (size_t)(ks - 1) * 512 + (size_t)(m0 + lrow) * HIDDEN + hlf * 16;

  const size_t k0 = (size_t)ks * 512;
  const bf16_t* pb0 = Wcat + ((size_t)(0 * HIDDEN + n0 + lrow)) * KCAT + k0 + hlf * 16;
  const bf16_t* pb1 = Wcat + ((size_t)(1 * HIDDEN + n0 + lrow)) * KCAT + k0 + hlf * 16;
  const bf16_t* pb2 = Wcat + ((size_t)(2 * HIDDEN + n0 + lrow)) * KCAT + k0 + hlf * 16;
  const bf16_t* pb3 = Wcat + ((size_t)(3 * HIDDEN + n0 + lrow)) * KCAT + k0 + hlf * 16;

  v8f acc0 = {}, acc1 = {}, acc2 = {}, acc3 = {};

  // -------- 2-deep software pipeline, fully unrolled (rotating regs) -------
  v16bf Ab[3], B0b[3], B1b[3], B2b[3], B3b[3];
  Ab[0] = ldA(pa, 0); B0b[0] = ldB(pb0, 0); B1b[0] = ldB(pb1, 0);
  B2b[0] = ldB(pb2, 0); B3b[0] = ldB(pb3, 0);
  Ab[1] = ldA(pa, 1); B0b[1] = ldB(pb0, 1); B1b[1] = ldB(pb1, 1);
  B2b[1] = ldB(pb2, 1); B3b[1] = ldB(pb3, 1);

#pragma unroll
  for (int c = 0; c < NCHUNK; ++c) {
    const int nc = c + 2;
    if (nc < NCHUNK) {
      const int s = nc % 3;
      Ab[s]  = ldA(pa, nc);
      B0b[s] = ldB(pb0, nc);
      B1b[s] = ldB(pb1, nc);
      B2b[s] = ldB(pb2, nc);
      B3b[s] = ldB(pb3, nc);
    }
    const int u = c % 3;
    acc0 = WMMA_BF16(Ab[u], B0b[u], acc0);
    acc1 = WMMA_BF16(Ab[u], B1b[u], acc1);
    acc2 = WMMA_BF16(Ab[u], B2b[u], acc2);
    acc3 = WMMA_BF16(Ab[u], B3b[u], acc3);
  }

  // -------- LDS reduction over K splits --------
  if (ks != 0) {
    float* base = &red[(size_t)(((ks - 1) * 4 + mq) * 4) * 256 + lane * 8];
    *(v8f*)(base + 0 * 256) = acc0;
    *(v8f*)(base + 1 * 256) = acc1;
    *(v8f*)(base + 2 * 256) = acc2;
    *(v8f*)(base + 3 * 256) = acc3;
  }
  __syncthreads();
  if (ks != 0) return;

#pragma unroll
  for (int s = 0; s < NSPLIT - 1; ++s) {
    const float* base = &red[(size_t)((s * 4 + mq) * 4) * 256 + lane * 8];
    acc0 += *(const v8f*)(base + 0 * 256);
    acc1 += *(const v8f*)(base + 1 * 256);
    acc2 += *(const v8f*)(base + 2 * 256);
    acc3 += *(const v8f*)(base + 3 * 256);
  }

  // -------- fused LSTM cell epilogue (PyTorch gate order i,f,g,o) --------
  const int col = n0 + lrow;
  const float bi = bias[col];
  const float bf = bias[HIDDEN + col];
  const float bg = bias[2 * HIDDEN + col];
  const float bo = bias[3 * HIDDEN + col];
#pragma unroll
  for (int r = 0; r < 8; ++r) {
    const int b = m0 + r + 8 * hlf; // C layout: M = r + 8*half, N = lane&15
    const size_t ci = (size_t)b * HIDDEN + col;
    float iv = sigm(acc0[r] + bi);
    float fv = sigm(acc1[r] + bf);
    float gv = tanhf(acc2[r] + bg);
    float ov = sigm(acc3[r] + bo);
    float cn = fv * cbuf[ci] + iv * gv;
    float hn = ov * tanhf(cn);
    cbuf[ci] = cn;
    bf16_t hb = f2bf(hn);
    hout[ci] = hb;
    Hall[((size_t)b * SEQ + t) * HIDDEN + col] = hb;
  }
}

// ---------------------------------------------------------------------------
// Batched FC: out[16384,1024] = Hall @ Wfc^T + b_fc   (bf16 WMMA, f32 out)
// Grid: (1024/64, 16384/64) x 128 threads; each wave: 16 rows x 64 cols.
// Fully unrolled 32-chunk K loop, immediate-offset loads, 2-deep pipeline.
// ---------------------------------------------------------------------------
__global__ void __launch_bounds__(128)
k_fc(const bf16_t* __restrict__ Hall, const bf16_t* __restrict__ Wfcb,
     const float* __restrict__ bfc, float* __restrict__ out) {
  const int lane = threadIdx.x & 31;
  const int wid  = __builtin_amdgcn_readfirstlane((int)threadIdx.x) >> 5; // scalar
  const int hlf  = lane >> 4;
  const int lrow = lane & 15;
  const size_t m0 = (size_t)blockIdx.y * 64 + (size_t)wid * 16;
  const int nb = blockIdx.x * 64;

  const bf16_t* pa  = Hall + (m0 + lrow) * HIDDEN + hlf * 16;
  const bf16_t* pb0 = Wfcb + (size_t)(nb + 0 * 16 + lrow) * HIDDEN + hlf * 16;
  const bf16_t* pb1 = Wfcb + (size_t)(nb + 1 * 16 + lrow) * HIDDEN + hlf * 16;
  const bf16_t* pb2 = Wfcb + (size_t)(nb + 2 * 16 + lrow) * HIDDEN + hlf * 16;
  const bf16_t* pb3 = Wfcb + (size_t)(nb + 3 * 16 + lrow) * HIDDEN + hlf * 16;

  v8f acc0 = {}, acc1 = {}, acc2 = {}, acc3 = {};

  v16bf Ab[3], B0b[3], B1b[3], B2b[3], B3b[3];
  Ab[0] = ldA(pa, 0); B0b[0] = ldB(pb0, 0); B1b[0] = ldB(pb1, 0);
  B2b[0] = ldB(pb2, 0); B3b[0] = ldB(pb3, 0);
  Ab[1] = ldA(pa, 1); B0b[1] = ldB(pb0, 1); B1b[1] = ldB(pb1, 1);
  B2b[1] = ldB(pb2, 1); B3b[1] = ldB(pb3, 1);

#pragma unroll
  for (int c = 0; c < HIDDEN / 32; ++c) {
    const int nc = c + 2;
    if (nc < HIDDEN / 32) {
      const int s = nc % 3;
      Ab[s]  = ldA(pa, nc);
      B0b[s] = ldB(pb0, nc);
      B1b[s] = ldB(pb1, nc);
      B2b[s] = ldB(pb2, nc);
      B3b[s] = ldB(pb3, nc);
      if ((nc & 3) == 0) __builtin_prefetch(pb0 + nc * 32 + 256, 0, 3);
    }
    const int u = c % 3;
    acc0 = WMMA_BF16(Ab[u], B0b[u], acc0);
    acc1 = WMMA_BF16(Ab[u], B1b[u], acc1);
    acc2 = WMMA_BF16(Ab[u], B2b[u], acc2);
    acc3 = WMMA_BF16(Ab[u], B3b[u], acc3);
  }

  const v8f accs[4] = {acc0, acc1, acc2, acc3};
#pragma unroll
  for (int nt = 0; nt < 4; ++nt) {
    const int col = nb + nt * 16 + lrow;
    const float bb = bfc[col];
#pragma unroll
    for (int r = 0; r < 8; ++r) {
      const size_t row = m0 + r + 8 * hlf;
      out[row * HIDDEN + col] = accs[nt][r] + bb;
    }
  }
}

// ---------------------------------------------------------------------------
extern "C" void kernel_launch(void* const* d_in, const int* in_sizes, int n_in,
                              void* d_out, int out_size, void* d_ws, size_t ws_size,
                              hipStream_t stream) {
  const float* emb = (const float*)d_in[0]; // [B,T,E]
  const float* Wih = (const float*)d_in[1]; // [4H,E]
  const float* Whh = (const float*)d_in[2]; // [4H,H]
  const float* bih = (const float*)d_in[3]; // [4H]
  const float* bhh = (const float*)d_in[4]; // [4H]
  const float* Wfc = (const float*)d_in[5]; // [H,H]
  const float* bfc = (const float*)d_in[6]; // [H]
  float* out = (float*)d_out;               // [B,T,H] fp32

  // Workspace carve-out (~65.4 MB total)
  size_t off = 0;
  char* ws = (char*)d_ws;
  auto sub = [&](size_t bytes) -> void* {
    void* p = (void*)(ws + off);
    off += (bytes + 255) & ~(size_t)255;
    return p;
  };
  bf16_t* Wcat = (bf16_t*)sub((size_t)GATES * KCAT * sizeof(bf16_t));        // 12.6 MB
  bf16_t* Wfcb = (bf16_t*)sub((size_t)HIDDEN * HIDDEN * sizeof(bf16_t));     //  2.0 MB
  float*  bias = (float*) sub((size_t)GATES * sizeof(float));                // 16 KB
  bf16_t* Xb   = (bf16_t*)sub((size_t)BATCH * SEQ * EMBED * sizeof(bf16_t)); // 16.8 MB
  bf16_t* h0   = (bf16_t*)sub((size_t)BATCH * HIDDEN * sizeof(bf16_t));      // 128 KB
  bf16_t* h1   = (bf16_t*)sub((size_t)BATCH * HIDDEN * sizeof(bf16_t));      // 128 KB
  float*  cbuf = (float*) sub((size_t)BATCH * HIDDEN * sizeof(float));       // 256 KB
  bf16_t* Hall = (bf16_t*)sub((size_t)BATCH * SEQ * HIDDEN * sizeof(bf16_t)); // 33.6 MB
  (void)ws_size; (void)in_sizes; (void)n_in; (void)out_size;

  // One-time conversions + state init (cheap, fully parallel)
  k_wcat<<<2048, 256, 0, stream>>>(Wih, Whh, Wcat);
  k_cvt_bf16<<<1024, 256, 0, stream>>>(Wfc, Wfcb, (size_t)HIDDEN * HIDDEN);
  k_bias<<<16, 256, 0, stream>>>(bih, bhh, bias);
  k_cvt_bf16<<<4096, 256, 0, stream>>>(emb, Xb, (size_t)BATCH * SEQ * EMBED);
  k_state_init<<<256, 256, 0, stream>>>(h0, cbuf);

  // Sequential recurrence: kernel-launch boundaries provide the global sync;
  // h ping-pongs so no block reads a buffer another block is writing.
  bf16_t* hp[2] = {h0, h1};
  for (int t = 0; t < SEQ; ++t) {
    k_lstm_step<<<64, 384, 0, stream>>>(Xb, Wcat, bias, hp[t & 1], hp[(t + 1) & 1],
                                        cbuf, Hall, t);
  }

  // Batched output projection over all (b,t)
  k_fc<<<dim3(HIDDEN / 64, (BATCH * SEQ) / 64), 128, 0, stream>>>(Hall, Wfcb, bfc, out);
}
